// SelfAttention_65481071403937
// MI455X (gfx1250) — matrix operand order repaired and checked by reading, not compiled, not run on hardware
//
#include <hip/hip_runtime.h>
#include <cstdint>

#define E     1024
#define HEADS 16
#define HD    64
#define BATCH 4
#define SEQ   2048
#define ROWS  (BATCH * SEQ)   // 8192

// softmax scale folded into base-2 domain: D^-0.5 * log2(e)
#define SCALE_LOG2E 0.18033688011112042f

typedef __attribute__((ext_vector_type(16))) _Float16 v16h;
typedef __attribute__((ext_vector_type(8)))  _Float16 v8h;
typedef __attribute__((ext_vector_type(8)))  float    v8f;

static __device__ __forceinline__ v16h cat8(v8h a, v8h b) {
    return __builtin_shufflevector(a, b, 0,1,2,3,4,5,6,7,8,9,10,11,12,13,14,15);
}

static __device__ __forceinline__ v8f wmma_f16(v16h a, v16h b, v8f c) {
    // D = A(16x32 f16) * B(32x16 f16) + C(16x16 f32)
    return __builtin_amdgcn_wmma_f32_16x16x32_f16(
        /*neg_a=*/false, a, /*neg_b=*/false, b,
        /*c_mod=*/(short)0, c, /*reuse_a=*/false, /*reuse_b=*/false);
}

// CDNA5 async DMA: per-lane 16B copy global -> LDS, tracked by ASYNCcnt.
static __device__ __forceinline__ void async_copy_b128(uint32_t lds_off, const void* gaddr) {
    asm volatile("global_load_async_to_lds_b128 %0, %1, off"
                 :: "v"(lds_off), "v"(gaddr) : "memory");
}
static __device__ __forceinline__ void wait_async0() {
    asm volatile("s_wait_asynccnt 0x0" ::: "memory");
}
static __device__ __forceinline__ void wait_async4() {   // keep newest stage in flight
    asm volatile("s_wait_asynccnt 0x4" ::: "memory");
}

// ---------------------------------------------------------------------------
// Kernel 1: Xh = f16(x), row-major.  (Also the GEMM A operand.)
// ---------------------------------------------------------------------------
__global__ __launch_bounds__(256) void pack_x_kernel(const float* __restrict__ x,
                                                     _Float16* __restrict__ xh) {
    const size_t i = ((size_t)blockIdx.x * 256 + threadIdx.x) * 8;
    float4 f0 = *reinterpret_cast<const float4*>(x + i);
    float4 f1 = *reinterpret_cast<const float4*>(x + i + 4);
    v8h h;
    h[0] = (_Float16)f0.x; h[1] = (_Float16)f0.y;
    h[2] = (_Float16)f0.z; h[3] = (_Float16)f0.w;
    h[4] = (_Float16)f1.x; h[5] = (_Float16)f1.y;
    h[6] = (_Float16)f1.z; h[7] = (_Float16)f1.w;
    *reinterpret_cast<v8h*>(xh + i) = h;
}

// ---------------------------------------------------------------------------
// Kernel 2: per-head transposed values: vT[(b*H+h)*HD + d][s] = f16(x[b,s,h*HD+d])
// ---------------------------------------------------------------------------
__global__ __launch_bounds__(256) void pack_vt_kernel(const float* __restrict__ x,
                                                      _Float16* __restrict__ vT) {
    __shared__ _Float16 lt[64 * 80];
    const int h  = blockIdx.x & 15;
    const int sb = (blockIdx.x >> 4) & 31;
    const int b  = blockIdx.x >> 9;
    const int tid = threadIdx.x;

    const int sl = tid >> 2;
    const int d0 = (tid & 3) * 16;
    const float* xp = x + (size_t)(b * SEQ + sb * 64 + sl) * E + h * HD + d0;
    #pragma unroll
    for (int j = 0; j < 4; ++j) {
        float4 f = *reinterpret_cast<const float4*>(xp + j * 4);
        lt[(d0 + j * 4 + 0) * 80 + sl] = (_Float16)f.x;
        lt[(d0 + j * 4 + 1) * 80 + sl] = (_Float16)f.y;
        lt[(d0 + j * 4 + 2) * 80 + sl] = (_Float16)f.z;
        lt[(d0 + j * 4 + 3) * 80 + sl] = (_Float16)f.w;
    }
    __syncthreads();
    const int dr = tid >> 2;
    const int c  = tid & 3;
    v16h vv = *reinterpret_cast<const v16h*>(&lt[dr * 80 + c * 16]);
    _Float16* op = vT + ((size_t)(b * HEADS + h) * HD + dr) * SEQ + sb * 64 + c * 16;
    *reinterpret_cast<v16h*>(op) = vv;
}

// ---------------------------------------------------------------------------
// Kernel 3: Wt[f][e] = f16(W[e][f])
// ---------------------------------------------------------------------------
__global__ __launch_bounds__(256) void pack_wt_kernel(const float* __restrict__ W,
                                                      _Float16* __restrict__ Wt) {
    __shared__ _Float16 lt[64 * 80];
    const int tf = blockIdx.x & 15;
    const int te = blockIdx.x >> 4;
    const int tid = threadIdx.x;

    const int er = tid >> 2;
    const int f0 = (tid & 3) * 16;
    const float* wp = W + (size_t)(te * 64 + er) * E + tf * 64 + f0;
    #pragma unroll
    for (int j = 0; j < 4; ++j) {
        float4 f = *reinterpret_cast<const float4*>(wp + j * 4);
        lt[(f0 + j * 4 + 0) * 80 + er] = (_Float16)f.x;
        lt[(f0 + j * 4 + 1) * 80 + er] = (_Float16)f.y;
        lt[(f0 + j * 4 + 2) * 80 + er] = (_Float16)f.z;
        lt[(f0 + j * 4 + 3) * 80 + er] = (_Float16)f.w;
    }
    __syncthreads();
    const int fr = tid >> 2;
    const int c  = tid & 3;
    v16h vv = *reinterpret_cast<const v16h*>(&lt[fr * 80 + c * 16]);
    *reinterpret_cast<v16h*>(Wt + (size_t)(tf * 64 + fr) * E + te * 64 + c * 16) = vv;
}

// ---------------------------------------------------------------------------
// Kernel 4: Out(f16) = Xh @ Wt^T, double-buffered async-to-LDS pipeline.
// 128x128 tile, 8 waves, each wave 32x64 (2x4 WMMA tiles), k-step 32.
// ---------------------------------------------------------------------------
static __device__ __forceinline__ void gemm_stage(const _Float16* __restrict__ Xh,
                                                  const _Float16* __restrict__ Wt,
                                                  int rowBase, int colBase, int k0,
                                                  uint32_t baseX, uint32_t baseW, int tid) {
    #pragma unroll
    for (int j = 0; j < 2; ++j) {
        const int idx = tid + j * 256;              // 0..511
        const int r = idx >> 2;                     // 0..127
        const int c = idx & 3;                      // 16B chunk
        const uint32_t loff = (uint32_t)(r * 96 + c * 16);
        async_copy_b128(baseX + loff, Xh + (size_t)(rowBase + r) * E + k0 + c * 8);
        async_copy_b128(baseW + loff, Wt + (size_t)(colBase + r) * E + k0 + c * 8);
    }
}

__global__ __launch_bounds__(256) void gemm_qk_kernel(const _Float16* __restrict__ Xh,
                                                      const _Float16* __restrict__ Wt,
                                                      _Float16* __restrict__ Out) {
    __shared__ _Float16 lX0[128 * 48], lX1[128 * 48];   // [row][k]
    __shared__ _Float16 lW0[128 * 48], lW1[128 * 48];   // [col][k]

    const int tid  = threadIdx.x;
    const int wave = tid >> 5;
    const int lane = tid & 31;
    const int l15  = lane & 15;
    const int hi   = lane >> 4;

    const int tileM = blockIdx.x >> 3;
    const int tileN = blockIdx.x & 7;
    const int rowBase = tileM * 128;
    const int colBase = tileN * 128;
    const int waveM = wave >> 1;
    const int waveN = wave & 1;

    const uint32_t baseX0 = (uint32_t)(uintptr_t)&lX0[0];
    const uint32_t baseX1 = (uint32_t)(uintptr_t)&lX1[0];
    const uint32_t baseW0 = (uint32_t)(uintptr_t)&lW0[0];
    const uint32_t baseW1 = (uint32_t)(uintptr_t)&lW1[0];

    v8f acc[2][4] = {};

    gemm_stage(Xh, Wt, rowBase, colBase, 0, baseX0, baseW0, tid);   // prologue

    for (int it = 0; it < 32; ++it) {
        const int cur = it & 1;
        if (it + 1 < 32) {
            // issue next stage into the other buffer, then drain only current
            gemm_stage(Xh, Wt, rowBase, colBase, (it + 1) * 32,
                       cur ? baseX0 : baseX1, cur ? baseW0 : baseW1, tid);
            wait_async4();
        } else {
            wait_async0();
        }
        __syncthreads();

        const _Float16* curX = cur ? lX1 : lX0;
        const _Float16* curW = cur ? lW1 : lW0;

        v16h a[2], b[4];
        #pragma unroll
        for (int mt = 0; mt < 2; ++mt) {
            const _Float16* p = &curX[(waveM * 32 + mt * 16 + l15) * 48];
            v8h lo = *reinterpret_cast<const v8h*>(p + hi * 8);
            v8h hh = *reinterpret_cast<const v8h*>(p + 16 + hi * 8);
            a[mt] = cat8(lo, hh);
        }
        #pragma unroll
        for (int nt = 0; nt < 4; ++nt) {
            const _Float16* p = &curW[(waveN * 64 + nt * 16 + l15) * 48];
            b[nt] = cat8(*reinterpret_cast<const v8h*>(p + hi * 8),
                         *reinterpret_cast<const v8h*>(p + 16 + hi * 8));
        }

        #pragma unroll
        for (int mt = 0; mt < 2; ++mt)
            #pragma unroll
            for (int nt = 0; nt < 4; ++nt)
                acc[mt][nt] = wmma_f16(a[mt], b[nt], acc[mt][nt]);

        __syncthreads();   // all waves done reading before this buffer is re-filled
    }

    #pragma unroll
    for (int mt = 0; mt < 2; ++mt)
        #pragma unroll
        for (int nt = 0; nt < 4; ++nt)
            #pragma unroll
            for (int r = 0; r < 8; ++r) {
                const int row = rowBase + waveM * 32 + mt * 16 + r + 8 * hi;
                const int col = colBase + waveN * 64 + nt * 16 + l15;
                Out[(size_t)row * E + col] = (_Float16)acc[mt][nt][r];
            }
}

// ---------------------------------------------------------------------------
// Kernel 5: flash attention, barrier-free inner loop, 64 keys per step.
// One block per (b, h, 128-query slab); 8 independent waves x 16 query rows.
// ---------------------------------------------------------------------------
__global__ __launch_bounds__(256) void attn_kernel(const _Float16* __restrict__ Q,
                                                   const _Float16* __restrict__ K,
                                                   const _Float16* __restrict__ vT,
                                                   float* __restrict__ Ctx) {
    __shared__ _Float16 pbuf[8 * 16 * 80];  // per-wave 16x64 P scratch (C->A relayout)

    const int tid  = threadIdx.x;
    const int wave = tid >> 5;
    const int lane = tid & 31;
    const int l15  = lane & 15;
    const int hi   = lane >> 4;

    const int qb = blockIdx.x & 15;
    const int h  = (blockIdx.x >> 4) & 15;
    const int b  = blockIdx.x >> 8;
    const int rowBase = b * SEQ + qb * 128;
    const int headCol = h * HD;
    const _Float16* vb = vT + (size_t)((b * HEADS + h) * HD) * SEQ;

    // Q A-fragments in registers (D=64 -> two K=32 chunks)
    v16h aQ[2];
    {
        const _Float16* qp = Q + (size_t)(rowBase + wave * 16 + l15) * E + headCol;
        #pragma unroll
        for (int c = 0; c < 2; ++c) {
            v8h lo = *reinterpret_cast<const v8h*>(qp + c * 32 + hi * 8);
            v8h hh = *reinterpret_cast<const v8h*>(qp + c * 32 + 16 + hi * 8);
            aQ[c] = cat8(lo, hh);
        }
    }

    float mrow[8], lrow[8];
    #pragma unroll
    for (int r = 0; r < 8; ++r) { mrow[r] = -1e30f; lrow[r] = 0.0f; }
    v8f accv[4] = {};
    _Float16* pb = &pbuf[wave * 16 * 80];

    for (int kb = 0; kb < SEQ; kb += 64) {
        // ---- scores: four 16x16 tiles over 64 keys (8 WMMAs)
        v8f s[4] = {};
        #pragma unroll
        for (int j = 0; j < 4; ++j) {
            const _Float16* kp = K + (size_t)(b * SEQ + kb + j * 16 + l15) * E
                                   + headCol + hi * 8;
            #pragma unroll
            for (int c = 0; c < 2; ++c) {
                v16h bk = cat8(*reinterpret_cast<const v8h*>(kp + c * 32),
                               *reinterpret_cast<const v8h*>(kp + c * 32 + 16));
                s[j] = wmma_f16(aQ[c], bk, s[j]);
            }
        }
        if (kb + 64 < SEQ) {   // hint next key block (global_prefetch_b8)
            __builtin_prefetch(K + (size_t)(b * SEQ + kb + 64 + l15) * E + headCol, 0, 1);
        }
        #pragma unroll
        for (int j = 0; j < 4; ++j) s[j] *= SCALE_LOG2E;   // base-2 softmax domain

        // ---- online softmax (row reductions across 16 lanes per half)
        #pragma unroll
        for (int r = 0; r < 8; ++r) {
            float t = fmaxf(fmaxf(s[0][r], s[1][r]), fmaxf(s[2][r], s[3][r]));
            #pragma unroll
            for (int msk = 1; msk < 16; msk <<= 1)
                t = fmaxf(t, __shfl_xor(t, msk, 32));
            const float nm   = fmaxf(mrow[r], t);
            const float corr = exp2f(mrow[r] - nm);   // v_exp_f32 is base-2
            float rs = 0.0f;
            #pragma unroll
            for (int j = 0; j < 4; ++j) {
                const float p = exp2f(s[j][r] - nm);
                s[j][r] = p;
                rs += p;
            }
            #pragma unroll
            for (int msk = 1; msk < 16; msk <<= 1)
                rs += __shfl_xor(rs, msk, 32);
            lrow[r] = lrow[r] * corr + rs;
            mrow[r] = nm;
            #pragma unroll
            for (int g = 0; g < 4; ++g) accv[g][r] *= corr;
        }

        // ---- P: C layout -> per-wave LDS -> two A fragments (K=64)
        #pragma unroll
        for (int r = 0; r < 8; ++r) {
            _Float16* dst = pb + (r + 8 * hi) * 80;
            #pragma unroll
            for (int j = 0; j < 4; ++j)
                dst[j * 16 + l15] = (_Float16)s[j][r];
        }
        asm volatile("s_wait_dscnt 0" ::: "memory");   // same-wave DS RAW
        v16h aP[2];
        {
            const _Float16* pp = pb + l15 * 80;
            #pragma unroll
            for (int c = 0; c < 2; ++c) {
                v8h lo = *reinterpret_cast<const v8h*>(pp + c * 32 + hi * 8);
                v8h hh = *reinterpret_cast<const v8h*>(pp + c * 32 + 16 + hi * 8);
                aP[c] = cat8(lo, hh);
            }
        }

        // ---- context += P . V  (B-frags straight from vT[d][keys], 8 WMMAs)
        #pragma unroll
        for (int c = 0; c < 2; ++c)
            #pragma unroll
            for (int g = 0; g < 4; ++g) {
                v16h bv = cat8(*reinterpret_cast<const v8h*>(
                    vb + (size_t)(g * 16 + l15) * SEQ + kb + c * 32 + hi * 8),
                    *reinterpret_cast<const v8h*>(
                    vb + (size_t)(g * 16 + l15) * SEQ + kb + c * 32 + 16 + hi * 8));
                accv[g] = wmma_f16(aP[c], bv, accv[g]);
            }
    }

    #pragma unroll
    for (int g = 0; g < 4; ++g)
        #pragma unroll
        for (int r = 0; r < 8; ++r) {
            const int row = rowBase + wave * 16 + r + 8 * hi;
            const int col = headCol + g * 16 + l15;
            Ctx[(size_t)row * E + col] = accv[g][r] / lrow[r];
        }
}

// ---------------------------------------------------------------------------
// Kernel 6: y = LN(context + x) * w + b        one block per row (E = 1024)
// ---------------------------------------------------------------------------
__global__ __launch_bounds__(256) void ln_kernel(const float* __restrict__ X,
                                                 const float* __restrict__ Ctx,
                                                 const float* __restrict__ w,
                                                 const float* __restrict__ bia,
                                                 float* __restrict__ Out) {
    __shared__ float rsum[8], rsq[8];
    const int row  = blockIdx.x;
    const int tid  = threadIdx.x;
    const int wave = tid >> 5;
    const int lane = tid & 31;

    const size_t base = (size_t)row * E + tid * 4;
    float4 xc = *reinterpret_cast<const float4*>(X + base);
    float4 cc = *reinterpret_cast<const float4*>(Ctx + base);
    float y0 = xc.x + cc.x, y1 = xc.y + cc.y, y2 = xc.z + cc.z, y3 = xc.w + cc.w;

    float s  = y0 + y1 + y2 + y3;
    float ss = y0 * y0 + y1 * y1 + y2 * y2 + y3 * y3;
    #pragma unroll
    for (int msk = 1; msk < 32; msk <<= 1) {
        s  += __shfl_xor(s,  msk, 32);
        ss += __shfl_xor(ss, msk, 32);
    }
    if (lane == 0) { rsum[wave] = s; rsq[wave] = ss; }
    __syncthreads();
    float ts = 0.f, tss = 0.f;
    #pragma unroll
    for (int i = 0; i < 8; ++i) { ts += rsum[i]; tss += rsq[i]; }

    const float mu   = ts * (1.0f / E);
    const float var  = tss * (1.0f / E) - mu * mu;
    const float rstd = rsqrtf(var + 1e-5f);

    float4 wv = *reinterpret_cast<const float4*>(w + tid * 4);
    float4 bv = *reinterpret_cast<const float4*>(bia + tid * 4);
    float4 o;
    o.x = (y0 - mu) * rstd * wv.x + bv.x;
    o.y = (y1 - mu) * rstd * wv.y + bv.y;
    o.z = (y2 - mu) * rstd * wv.z + bv.z;
    o.w = (y3 - mu) * rstd * wv.w + bv.w;
    *reinterpret_cast<float4*>(Out + base) = o;
}

// ---------------------------------------------------------------------------
// Launch
// ---------------------------------------------------------------------------
extern "C" void kernel_launch(void* const* d_in, const int* in_sizes, int n_in,
                              void* d_out, int out_size, void* d_ws, size_t ws_size,
                              hipStream_t stream) {
    const float* Wr  = (const float*)d_in[0];   // rotation_params (E,E)
    const float* We  = (const float*)d_in[1];   // entangle_params (E,E)
    const float* X   = (const float*)d_in[2];   // inputs (B,S,E)
    const float* lnw = (const float*)d_in[3];
    const float* lnb = (const float*)d_in[4];
    float* Out = (float*)d_out;

    const size_t NE = (size_t)ROWS * E;         // 8,388,608 elements
    _Float16* xh   = (_Float16*)d_ws;           // 16 MB  f16(x)
    _Float16* vTb  = xh   + NE;                 // 16 MB  per-head transposed values
    _Float16* qbuf = vTb  + NE;                 // 16 MB
    _Float16* kbuf = qbuf + NE;                 // 16 MB
    _Float16* wtA  = kbuf + NE;                 //  2 MB
    _Float16* wtB  = wtA  + (size_t)E * E;      //  2 MB
    float*    ctx  = (float*)(wtB + (size_t)E * E);  // 32 MB

    pack_x_kernel <<<(int)(NE / (256 * 8)), 256, 0, stream>>>(X, xh);
    pack_vt_kernel<<<BATCH * (SEQ / 64) * HEADS, 256, 0, stream>>>(X, vTb);
    pack_wt_kernel<<<(E / 64) * (E / 64), 256, 0, stream>>>(Wr, wtA);
    pack_wt_kernel<<<(E / 64) * (E / 64), 256, 0, stream>>>(We, wtB);
    gemm_qk_kernel<<<(ROWS / 128) * (E / 128), 256, 0, stream>>>(xh, wtA, qbuf);
    gemm_qk_kernel<<<(ROWS / 128) * (E / 128), 256, 0, stream>>>(xh, wtB, kbuf);
    attn_kernel   <<<BATCH * HEADS * (SEQ / 128), 256, 0, stream>>>(qbuf, kbuf, vTb, ctx);
    ln_kernel     <<<ROWS, 256, 0, stream>>>(X, ctx, lnw, lnb, Out);
}
